// MLPMHA_56023553409274
// MI455X (gfx1250) — matrix operand order repaired and checked by reading, not compiled, not run on hardware
//
#include <hip/hip_runtime.h>
#include <stdint.h>

typedef __attribute__((ext_vector_type(16))) _Float16 v16h;
typedef __attribute__((ext_vector_type(8)))  _Float16 v8h;
typedef __attribute__((ext_vector_type(8)))  float    v8f;
typedef __attribute__((ext_vector_type(4)))  unsigned int v4u;
typedef __attribute__((ext_vector_type(8)))  unsigned int v8u;

#define B_   4
#define T_   2048
#define C_   1024
#define QH_  4
#define D_   256
#define S_   4096                // 4*C keys/values
#define NTOK (B_ * T_)           // 8192 token rows
#define NTILE (NTOK * QH_ / 16)  // 2048 (b,h,t-tile) work tiles
#define NCHUNK (S_ / 32)         // 128 S-chunks

// Dynamic LDS layout in halves.
#define KBUF0 0                  // 32x256 K chunk, double buffered
#define KBUF1 8192
#define VBUF0 16384              // 256x32 V^T chunk, double buffered
#define VBUF1 24576
#define PBUF  32768              // + wid*512  (per-wave 16x32 P relayout)
#define QBUF  36864              // + wid*4096 (per-wave 16x256 q tile)
#define SMEM_HALVES (QBUF + 8 * 4096)
#define SMEM_BYTES  (SMEM_HALVES * 2)    // 139264 B

// ---------------------------------------------------------------------------
// CDNA5 helpers
// ---------------------------------------------------------------------------
__device__ __forceinline__ v8f wmma16(v16h a, v16h b, v8f c) {
  return __builtin_amdgcn_wmma_f32_16x16x32_f16(false, a, false, b, (short)0,
                                                c, false, false);
}

__device__ __forceinline__ v8f zero8() {
  v8f z = {0.f, 0.f, 0.f, 0.f, 0.f, 0.f, 0.f, 0.f};
  return z;
}

// Async global -> LDS copy of 16 bytes per lane (ASYNCcnt-tracked, GV mode).
__device__ __forceinline__ void async_b128(_Float16* lds_dst, const _Float16* gsrc) {
  unsigned l = (unsigned)(uintptr_t)lds_dst;                 // LDS byte offset
  unsigned long long g = (unsigned long long)(uintptr_t)gsrc;
  asm volatile("global_load_async_to_lds_b128 %0, %1, off"
               :: "v"(l), "v"(g) : "memory");
}

__device__ __forceinline__ void wait_async0() {
  asm volatile("s_wait_asynccnt 0x0" ::: "memory");
}

// Tensor Data Mover: 2D tile load (TENSORcnt-tracked). Tile = 32 rows x 256
// f16 elements with row pitch 256 (i.e. one contiguous 16KB K chunk).
// D# per CDNA5 ISA: group0 = {count|flags, lds_addr,
// gaddr_lo, gaddr_hi|type=2}; group1 = {data_size, dims/tile/strides}.
__device__ __forceinline__ void tdm_load_k_chunk(unsigned lds_byte_off,
                                                 const _Float16* gsrc) {
  const unsigned long long ga = (unsigned long long)(uintptr_t)gsrc;
  v4u g0;
  g0.x = 1u;                                                 // count=1 (valid)
  g0.y = lds_byte_off;                                       // LDS byte addr
  g0.z = (unsigned)ga;                                       // gaddr[31:0]
  g0.w = ((unsigned)(ga >> 32) & 0x01FFFFFFu) | (2u << 30);  // gaddr[56:32]|type=2
  v8u g1;
  g1.s0 = 1u << 16;              // workgroup_mask=0, data_size=1 (2 bytes)
  g1.s1 = (unsigned)D_ << 16;    // tensor_dim0 = 256   (bits 79:48)
  g1.s2 = (unsigned)S_ << 16;    // tensor_dim1 = 4096  (bits 111:80)
  g1.s3 = (unsigned)D_ << 16;    // tile_dim0   = 256   (bits 127:112)
  g1.s4 = 32u;                   // tile_dim1 = 32, tile_dim2 = 0
  g1.s5 = (unsigned)D_;          // tensor_dim0_stride = 256 (bits 207:160)
  g1.s6 = 0u;                    // tensor_dim1_stride unused for 2D
  g1.s7 = 0u;
  asm volatile("tensor_load_to_lds %0, %1" :: "s"(g0), "s"(g1) : "memory");
}

// A-fragment (16x32 f16, row-major source). contiguous k order (not the ISA table order), matched by load_bt16:
//   lane L (row = L&15, sel = L>>4) holds K = sel*16..+15 (contiguous).
__device__ __forceinline__ v16h load_a16(const _Float16* p, int stride, int lane) {
  const int r = lane & 15;
  const int sel = lane >> 4;
  const _Float16* base = p + r * stride + sel * 16;
  v8h lo = *(const v8h*)(base);
  v8h hi = *(const v8h*)(base + 8);
  v16h a;
#pragma unroll
  for (int i = 0; i < 8; ++i) { a[i] = lo[i]; a[i + 8] = hi[i]; }
  return a;
}

// B-fragment (32x16 f16) from B^T stored row-major (n-major, K inner):
//   lane L holds column n = L&15, K = (L>>4)*16 + j -> 32 contiguous bytes.
__device__ __forceinline__ v16h load_bt16(const _Float16* p, int stride, int lane) {
  const _Float16* base = p + (lane & 15) * stride + (lane >> 4) * 16;
  return *(const v16h*)base;
}

// ---------------------------------------------------------------------------
// Prep kernels
// ---------------------------------------------------------------------------
__global__ void __launch_bounds__(256)
ln_to_f16_kernel(const float* __restrict__ x, const float* __restrict__ g,
                 const float* __restrict__ be, _Float16* __restrict__ x16) {
  __shared__ float red[256];
  const int row = blockIdx.x;
  const int t = threadIdx.x;
  const float* xr = x + (size_t)row * C_;
  float v0 = xr[t], v1 = xr[t + 256], v2 = xr[t + 512], v3 = xr[t + 768];
  red[t] = v0 + v1 + v2 + v3;
  __syncthreads();
#pragma unroll
  for (int off = 128; off > 0; off >>= 1) {
    if (t < off) red[t] += red[t + off];
    __syncthreads();
  }
  const float mean = red[0] * (1.f / C_);
  __syncthreads();
  const float d0 = v0 - mean, d1 = v1 - mean, d2 = v2 - mean, d3 = v3 - mean;
  red[t] = d0 * d0 + d1 * d1 + d2 * d2 + d3 * d3;
  __syncthreads();
#pragma unroll
  for (int off = 128; off > 0; off >>= 1) {
    if (t < off) red[t] += red[t + off];
    __syncthreads();
  }
  const float rstd = rsqrtf(red[0] * (1.f / C_) + 1e-5f);
  _Float16* o = x16 + (size_t)row * C_;
  o[t]       = (_Float16)(d0 * rstd * g[t]       + be[t]);
  o[t + 256] = (_Float16)(d1 * rstd * g[t + 256] + be[t + 256]);
  o[t + 512] = (_Float16)(d2 * rstd * g[t + 512] + be[t + 512]);
  o[t + 768] = (_Float16)(d3 * rstd * g[t + 768] + be[t + 768]);
}

// K = w_fc.reshape(4096,256): identical flat layout -> straight f16 convert.
__global__ void __launch_bounds__(256)
conv_k_kernel(const float* __restrict__ w_fc, _Float16* __restrict__ k16) {
  const int i = blockIdx.x * 256 + threadIdx.x;
  k16[i] = (_Float16)w_fc[i];
}

// vT16[d][s] = V[s][d], V = w_proj^T.reshape(4096,256)
//   V.flat[s*256+d] = w_proj[((s%4)*256 + d) * 1024 + s/4]
__global__ void __launch_bounds__(256)
conv_v_kernel(const float* __restrict__ w_proj, _Float16* __restrict__ vT16) {
  const int i = blockIdx.x * 256 + threadIdx.x;
  const int d = i >> 12;      // / S_
  const int s = i & (S_ - 1);
  vT16[i] = (_Float16)w_proj[(size_t)((s & 3) * 256 + d) * C_ + (s >> 2)];
}

// ---------------------------------------------------------------------------
// Main kernel: block = 8 waves, each wave = 16 tokens of one head.
// K chunks staged by the TDM (tensor_load_to_lds, wave 0), V chunks by async
// global->LDS DMA (all waves), double-buffered; q tile lives in LDS.
// ---------------------------------------------------------------------------
__global__ void __launch_bounds__(256, 1)
mha_wmma_kernel(const _Float16* __restrict__ x16,
                const _Float16* __restrict__ k16,
                const _Float16* __restrict__ vT16,
                const float* __restrict__ residual,
                float* __restrict__ out) {
  extern __shared__ _Float16 smem[];
  const int tid  = threadIdx.x;
  const int lane = tid & 31;
  const int wid  = tid >> 5;
  const int tile = blockIdx.x * 8 + wid;        // 0..NTILE-1
  const int b  = tile >> 9;                     // / (QH_ * T_/16)
  const int rr = tile & 511;
  const int h  = rr >> 7;                       // / (T_/16)
  const int t0 = (rr & 127) << 4;
  const int row0 = b * T_ + t0;
  const int col0 = h * D_;
  _Float16* Pw = smem + PBUF + wid * 512;
  _Float16* Qw = smem + QBUF + wid * 4096;

  // ---- stage this wave's q tile (16 x 256 f16) into LDS via async DMA ----
  {
    const _Float16* qg = x16 + (size_t)row0 * C_ + col0;
#pragma unroll
    for (int r = 0; r < 16; ++r) {
      const int id = r * 32 + lane;       // 512 x 16B chunks
      const int qrow = id >> 5;
      const int off = (id & 31) * 8;
      async_b128(Qw + qrow * D_ + off, qg + (size_t)qrow * C_ + off);
    }
  }

  v8f acc[16];
#pragma unroll
  for (int i = 0; i < 16; ++i) acc[i] = zero8();

  const int m0 = (lane >> 4) * 8;
  const int nn = lane & 15;

  // V chunk async stage: Vlds[256][32] <- vT16[d][chunk*32 + s], pitch S_
  auto issue_v = [&](int chunk, int buf) {
    const _Float16* vg = vT16 + chunk * 32;
    _Float16* vb = smem + (buf ? VBUF1 : VBUF0);
#pragma unroll
    for (int r = 0; r < 4; ++r) {
      const int idx = r * 256 + tid;
      const int d = idx >> 2, sub = (idx & 3) * 8;
      async_b128(vb + d * 32 + sub, vg + (size_t)d * S_ + sub);
    }
  };
  // K chunk TDM stage: one descriptor, contiguous 16KB tile (wave 0 only)
  auto issue_k = [&](int chunk, int buf) {
    if (wid == 0)
      tdm_load_k_chunk((unsigned)((buf ? KBUF1 : KBUF0) * 2 +
                                  (unsigned)(uintptr_t)smem),
                       k16 + (size_t)chunk * (32 * D_));
  };

  issue_k(0, 0);
  issue_v(0, 0);

  for (int sc = 0; sc < NCHUNK; ++sc) {
    const int cur = sc & 1;
    if (wid == 0) __builtin_amdgcn_s_wait_tensorcnt(0);  // K chunk landed
    wait_async0();                                       // my V/q copies landed
    __syncthreads();                                     // block-wide ready
    if (sc + 1 < NCHUNK) {
      issue_k(sc + 1, cur ^ 1);
      issue_v(sc + 1, cur ^ 1);
    }

    const _Float16* Kb = smem + (cur ? KBUF1 : KBUF0);
    const _Float16* Vb = smem + (cur ? VBUF1 : VBUF0);

    // ---- scores: two 16x16 tiles over K=256 (q + K from LDS) ----
    v8f c0 = zero8();
    v8f c1 = zero8();
#pragma unroll
    for (int kc = 0; kc < 8; ++kc) {
      v16h qa  = load_a16(Qw + kc * 32, D_, lane);
      v16h bk0 = load_bt16(Kb + kc * 32, D_, lane);
      v16h bk1 = load_bt16(Kb + 16 * D_ + kc * 32, D_, lane);
      c0 = wmma16(qa, bk0, c0);
      c1 = wmma16(qa, bk1, c1);
    }
    // ---- relu^2 -> f16, C->A relayout via per-wave LDS ----
#pragma unroll
    for (int j = 0; j < 8; ++j) {
      float f0 = c0[j] > 0.f ? c0[j] : 0.f;
      float f1 = c1[j] > 0.f ? c1[j] : 0.f;
      Pw[(m0 + j) * 32 + nn]      = (_Float16)(f0 * f0);
      Pw[(m0 + j) * 32 + nn + 16] = (_Float16)(f1 * f1);
    }
    asm volatile("s_wait_dscnt 0x0" ::: "memory");  // same-wave LDS RAW
    v16h pf = load_a16(Pw, 32, lane);
    // ---- P(16x32) x V(32x256): 16 output d-tiles (from LDS) ----
#pragma unroll
    for (int dt = 0; dt < 16; ++dt) {
      v16h vb = load_bt16(Vb + dt * 16 * 32, 32, lane);
      acc[dt] = wmma16(pf, vb, acc[dt]);
    }
    asm volatile("" ::: "memory");  // keep LDS WAR order across iterations
  }

  // ---- epilogue: residual add, fp32 store ----
  const size_t obase = (size_t)row0 * C_ + col0;
#pragma unroll
  for (int dt = 0; dt < 16; ++dt) {
#pragma unroll
    for (int j = 0; j < 8; ++j) {
      const size_t idx = obase + (size_t)(m0 + j) * C_ + dt * 16 + nn;
      out[idx] = residual[idx] + acc[dt][j];
    }
  }
}

// ---------------------------------------------------------------------------
extern "C" void kernel_launch(void* const* d_in, const int* in_sizes, int n_in,
                              void* d_out, int out_size, void* d_ws, size_t ws_size,
                              hipStream_t stream) {
  const float* residual = (const float*)d_in[0];
  const float* w_fc     = (const float*)d_in[1];
  const float* w_proj   = (const float*)d_in[2];
  const float* ln_g     = (const float*)d_in[3];
  const float* ln_b     = (const float*)d_in[4];
  float* out = (float*)d_out;

  // Workspace layout (f16): x16 16MB | k16 2MB | vT16 2MB
  _Float16* x16 = (_Float16*)d_ws;
  _Float16* k16 = x16 + (size_t)NTOK * C_;
  _Float16* vT  = k16 + (size_t)S_ * D_;

  ln_to_f16_kernel<<<NTOK, 256, 0, stream>>>(residual, ln_g, ln_b, x16);
  conv_k_kernel<<<(C_ * C_) / 256, 256, 0, stream>>>(w_fc, k16);
  conv_v_kernel<<<(D_ * S_) / 256, 256, 0, stream>>>(w_proj, vT);
  mha_wmma_kernel<<<NTILE / 8, 256, SMEM_BYTES, stream>>>(x16, k16, vT,
                                                          residual, out);
}
